// PlasmidLMSparseMoE_17257178595381
// MI455X (gfx1250) — matrix operand order repaired and checked by reading, not compiled, not run on hardware
//
#include <hip/hip_runtime.h>
#include <math.h>

// ---------------- problem constants (match reference) ----------------
#define T_TOK   8192            // B*S = 4*2048
#define H_DIM   1024
#define F_DIM   2048
#define NEXP    8
#define TOPK    2
#define NSLOT   (T_TOK * TOPK)          // 16384
#define MAXSLOTS (NSLOT + NEXP * 16)    // 16512 (per-expert pad to 16)
#define MAXTILES (NSLOT / 16 + NEXP)    // 1032 fixed grid for graph capture

typedef __bf16 bf16;
typedef __attribute__((ext_vector_type(16))) __bf16 v16bf;
typedef __attribute__((ext_vector_type(8)))  __bf16 v8bf;
typedef __attribute__((ext_vector_type(4)))  __bf16 v4bf;
typedef __attribute__((ext_vector_type(8)))  float  v8f;

__device__ __forceinline__ bf16 f2bf(float x) { return (bf16)x; }

// Build a 16-element bf16 WMMA fragment from two contiguous 16-byte chunks.
__device__ __forceinline__ v16bf frag_ld(const bf16* p0, const bf16* p1) {
    v8bf lo = *(const v8bf*)p0;
    v8bf hi = *(const v8bf*)p1;
    return __builtin_shufflevector(lo, hi, 0,1,2,3,4,5,6,7,8,9,10,11,12,13,14,15);
}

// ---------------- kernel 1: init scratch ----------------
__global__ void __launch_bounds__(256)
init_kernel(int* __restrict__ slot2tok, int* __restrict__ counts) {
    int i = blockIdx.x * 256 + threadIdx.x;
    if (i < MAXSLOTS) slot2tok[i] = 0;      // padded rows gather token 0 (never read back)
    if (i < NEXP)     counts[i]  = 0;
}

// ---------------- kernel 2: hidden fp32 -> bf16 ----------------
__global__ void __launch_bounds__(256)
cvtx_kernel(const float* __restrict__ X, bf16* __restrict__ Xbf) {
    int gid = blockIdx.x * 256 + threadIdx.x;          // groups of 4
    float4 v = ((const float4*)X)[gid];
    v4bf o = { f2bf(v.x), f2bf(v.y), f2bf(v.z), f2bf(v.w) };
    ((v4bf*)Xbf)[gid] = o;
}

// ------- kernel 3: per-expert transpose + fp32->bf16 (LDS-tiled) -------
// src: [E][R][C] fp32  ->  dst: [E][C][R] bf16
__global__ void __launch_bounds__(256)
transpose_cvt_kernel(const float* __restrict__ src, bf16* __restrict__ dst,
                     int R, int C) {
    __shared__ float tile[32][33];
    const size_t slice = (size_t)blockIdx.z * (size_t)R * (size_t)C;
    const int c0 = blockIdx.x * 32, r0 = blockIdx.y * 32;
    const int tx = threadIdx.x, ty = threadIdx.y;      // 32 x 8
#pragma unroll
    for (int i = 0; i < 32; i += 8)
        tile[ty + i][tx] = src[slice + (size_t)(r0 + ty + i) * C + c0 + tx];
    __syncthreads();
#pragma unroll
    for (int i = 0; i < 32; i += 8)
        dst[slice + (size_t)(c0 + ty + i) * R + r0 + tx] = f2bf(tile[tx][ty + i]);
}

// ---------------- kernel 4: router (one wave32 per token) ----------------
__global__ void __launch_bounds__(256)
router_kernel(const float* __restrict__ X, const float* __restrict__ Wr,
              float* __restrict__ probRow, int* __restrict__ eidx,
              float* __restrict__ tokw, int* __restrict__ counts) {
    const int wave = threadIdx.x >> 5, lane = threadIdx.x & 31;
    const int t = blockIdx.x * 8 + wave;
    if (t >= T_TOK) return;

    float p[NEXP];
#pragma unroll
    for (int e = 0; e < NEXP; ++e) p[e] = 0.f;

    for (int h = lane; h < H_DIM; h += 32) {
        float x = X[(size_t)t * H_DIM + h];
        const float4* w4 = (const float4*)(Wr + (size_t)h * NEXP);
        float4 a = w4[0], b = w4[1];
        p[0] += x * a.x; p[1] += x * a.y; p[2] += x * a.z; p[3] += x * a.w;
        p[4] += x * b.x; p[5] += x * b.y; p[6] += x * b.z; p[7] += x * b.w;
    }
#pragma unroll
    for (int off = 16; off > 0; off >>= 1)
#pragma unroll
        for (int e = 0; e < NEXP; ++e) p[e] += __shfl_xor(p[e], off, 32);

    if (lane == 0) {
        float m = p[0];
#pragma unroll
        for (int e = 1; e < NEXP; ++e) m = fmaxf(m, p[e]);
        float s = 0.f;
#pragma unroll
        for (int e = 0; e < NEXP; ++e) { p[e] = expf(p[e] - m); s += p[e]; }
        float inv = 1.f / s;
#pragma unroll
        for (int e = 0; e < NEXP; ++e) { p[e] *= inv; probRow[(size_t)t * NEXP + e] = p[e]; }
        int e0 = 0;
#pragma unroll
        for (int e = 1; e < NEXP; ++e) if (p[e] > p[e0]) e0 = e;
        int e1 = (e0 == 0) ? 1 : 0;
#pragma unroll
        for (int e = 0; e < NEXP; ++e) if (e != e0 && p[e] > p[e1]) e1 = e;
        float wsum = p[e0] + p[e1];
        eidx[2 * t]     = e0;  tokw[2 * t]     = p[e0] / wsum;
        eidx[2 * t + 1] = e1;  tokw[2 * t + 1] = p[e1] / wsum;
        atomicAdd(&counts[e0], 1);
        atomicAdd(&counts[e1], 1);
    }
}

// ------- kernel 5: aux loss + padded scan + tile map (single block) -------
__global__ void __launch_bounds__(256)
setup_kernel(const float* __restrict__ probRow, const int* __restrict__ counts,
             int* __restrict__ cursors, int* __restrict__ tileExpert,
             int* __restrict__ tileSlot, int* __restrict__ tileValid,
             float* __restrict__ aux_out) {
    __shared__ float sh[NEXP][256];
    const int tid = threadIdx.x;
    float s[NEXP];
#pragma unroll
    for (int e = 0; e < NEXP; ++e) s[e] = 0.f;
    for (int t = tid; t < T_TOK; t += 256)
#pragma unroll
        for (int e = 0; e < NEXP; ++e) s[e] += probRow[(size_t)t * NEXP + e];
#pragma unroll
    for (int e = 0; e < NEXP; ++e) sh[e][tid] = s[e];
    __syncthreads();
    if (tid < NEXP) {                       // fixed-order (deterministic) reduce
        float tot = 0.f;
        for (int i = 0; i < 256; ++i) tot += sh[tid][i];
        sh[tid][0] = tot;
    }
    __syncthreads();
    if (tid == 0) {
        int off = 0, b = 0;
        for (int e = 0; e < NEXP; ++e) {
            int c = counts[e];
            int tiles = (c + 15) >> 4;
            cursors[e] = off;
            for (int ti = 0; ti < tiles; ++ti) {
                tileExpert[b] = e;
                tileSlot[b]   = off + 16 * ti;
                int v = c - 16 * ti;
                tileValid[b]  = v > 16 ? 16 : v;
                ++b;
            }
            off += tiles * 16;
        }
        for (; b < MAXTILES; ++b) tileExpert[b] = -1;
        float aux = 0.f;
        for (int e = 0; e < NEXP; ++e)
            aux += ((float)counts[e] / (float)(T_TOK * TOPK)) * (sh[e][0] / (float)T_TOK);
        aux_out[0] = aux * (float)NEXP;
    }
}

// ---------------- kernel 6: scatter tokens into expert segments ----------------
__global__ void __launch_bounds__(256)
scatter_kernel(const int* __restrict__ eidx, int* __restrict__ cursors,
               int* __restrict__ slot2tok, int* __restrict__ tok2slot) {
    int t = blockIdx.x * 256 + threadIdx.x;
    if (t >= T_TOK) return;
#pragma unroll
    for (int k = 0; k < TOPK; ++k) {
        int e = eidx[2 * t + k];
        int pos = atomicAdd(&cursors[e], 1);
        slot2tok[pos] = t;
        tok2slot[2 * t + k] = pos;
    }
}

// ------------- kernel 7: fused up-GEMM -> GeLU -> down-GEMM (WMMA) -------------
// One block = 16 slots (rows) x full H output. 8 waves, each owns 128 output cols.
// WupT: [E][F][H] bf16 (K-contiguous rows), WdnT: [E][H][F] bf16.
__global__ void __launch_bounds__(256)
moe_gemm_kernel(const bf16* __restrict__ Xbf, const bf16* __restrict__ WupT,
                const bf16* __restrict__ WdnT, bf16* __restrict__ Ybf,
                const int* __restrict__ slot2tok, const int* __restrict__ tileExpert,
                const int* __restrict__ tileSlot, const int* __restrict__ tileValid) {
    __shared__ __align__(16) bf16 Xs[16][H_DIM];   // 32 KB token tile
    __shared__ __align__(16) bf16 Gs[16][128];     // 4 KB gelu(up) staging

    const int bid = blockIdx.x;
    const int e = tileExpert[bid];
    if (e < 0) return;
    const int slot0 = tileSlot[bid];
    const int valid = tileValid[bid];

    const int tid = threadIdx.x;
    {   // gather 16 rows of Xbf into LDS via async global->LDS (no VGPR roundtrip):
        // 16 threads/row, 8 x 16B per thread, tracked on ASYNCcnt.
        const int r = tid >> 4, part = tid & 15;
        const int tok = slot2tok[slot0 + r];
        const bf16* src = Xbf + (size_t)tok * H_DIM + part * 64;   // 64 bf16 = 128 B
        unsigned int lds_off = (unsigned int)(size_t)(&Xs[r][part * 64]);
#pragma unroll
        for (int i = 0; i < 8; ++i) {
            unsigned long long ga = (unsigned long long)(size_t)src + (unsigned)(i * 16);
            unsigned int lo = lds_off + (unsigned)(i * 16);
            asm volatile("global_load_async_to_lds_b128 %0, %1, off"
                         :: "v"(lo), "v"(ga) : "memory");
        }
        asm volatile("s_wait_asynccnt 0x0" ::: "memory");
    }
    __syncthreads();

    const int wave = tid >> 5;
    const int lane = tid & 31;
    const int half = lane >> 4;     // ISA A/B fragment half-wave select
    const int l16  = lane & 15;

    v8f zero = {};
    v8f acc[8];
#pragma unroll
    for (int i = 0; i < 8; ++i) acc[i] = zero;

    const bf16* wupE = WupT + (size_t)e * F_DIM * H_DIM;
    const bf16* wdnE = WdnT + (size_t)e * H_DIM * F_DIM;

    for (int c = 0; c < F_DIM / 128; ++c) {
        // ---- phase 1: U(16x16) for this wave's 16 F-columns, K over full H ----
        const int f = c * 128 + wave * 16 + l16;                 // B lane column
        const bf16* brow = wupE + (size_t)f * H_DIM + 16 * half; // K-contiguous
        if (c + 1 < F_DIM / 128)                                 // warm next chunk's B rows
            __builtin_prefetch(brow + (size_t)128 * H_DIM, 0, 0);
        v8f u = zero;
#pragma unroll 4
        for (int k0 = 0; k0 < H_DIM; k0 += 32) {
            v16bf a = frag_ld(&Xs[l16][k0 + 16 * half], &Xs[l16][k0 + 16 * half + 8]);
            v16bf b = frag_ld(brow + k0, brow + k0 + 8);
            u = __builtin_amdgcn_wmma_f32_16x16x32_bf16(false, a, false, b,
                                                        (short)0, u, false, false);
        }
        // exact-erf GeLU, bf16, stage to LDS (C layout: row = v + 8*half)
#pragma unroll
        for (int v = 0; v < 8; ++v) {
            float x = u[v];
            float g = 0.5f * x * (1.0f + erff(x * 0.70710678118654752f));
            Gs[v + 8 * half][wave * 16 + l16] = f2bf(g);
        }
        __syncthreads();
        // ---- phase 2: acc(16x128) += G(16x128) @ Wdn[f-chunk, wave's 128 cols] ----
#pragma unroll
        for (int ks = 0; ks < 4; ++ks) {
            const int k0 = ks * 32;
            v16bf a = frag_ld(&Gs[l16][k0 + 16 * half], &Gs[l16][k0 + 16 * half + 8]);
#pragma unroll
            for (int nt = 0; nt < 8; ++nt) {
                const int hcol = wave * 128 + nt * 16 + l16;
                const bf16* bp = wdnE + (size_t)hcol * F_DIM + c * 128 + k0 + 16 * half;
                v16bf b = frag_ld(bp, bp + 8);
                acc[nt] = __builtin_amdgcn_wmma_f32_16x16x32_bf16(false, a, false, b,
                                                                  (short)0, acc[nt],
                                                                  false, false);
            }
        }
        __syncthreads();
    }
    // ---- store Y rows (bf16), predicated on valid rows ----
#pragma unroll
    for (int nt = 0; nt < 8; ++nt) {
        const int hcol = wave * 128 + nt * 16 + l16;
#pragma unroll
        for (int v = 0; v < 8; ++v) {
            const int m = v + 8 * half;
            if (m < valid)
                Ybf[(size_t)(slot0 + m) * H_DIM + hcol] = f2bf(acc[nt][v]);
        }
    }
}

// ---------------- kernel 8: deterministic weighted combine ----------------
__global__ void __launch_bounds__(256)
combine_kernel(const bf16* __restrict__ Ybf, const int* __restrict__ tok2slot,
               const float* __restrict__ tokw, float* __restrict__ out) {
    int gid = blockIdx.x * 256 + threadIdx.x;       // groups of 4 along H
    int t  = gid / (H_DIM / 4);
    int h4 = (gid % (H_DIM / 4)) * 4;
    int s0 = tok2slot[2 * t], s1 = tok2slot[2 * t + 1];
    float w0 = tokw[2 * t],  w1 = tokw[2 * t + 1];
    v4bf a = *(const v4bf*)(Ybf + (size_t)s0 * H_DIM + h4);
    v4bf b = *(const v4bf*)(Ybf + (size_t)s1 * H_DIM + h4);
    float4 o;
    o.x = w0 * (float)a[0] + w1 * (float)b[0];
    o.y = w0 * (float)a[1] + w1 * (float)b[1];
    o.z = w0 * (float)a[2] + w1 * (float)b[2];
    o.w = w0 * (float)a[3] + w1 * (float)b[3];
    ((float4*)out)[gid] = o;
}

// ---------------- host-side launch ----------------
extern "C" void kernel_launch(void* const* d_in, const int* in_sizes, int n_in,
                              void* d_out, int out_size, void* d_ws, size_t ws_size,
                              hipStream_t stream) {
    (void)in_sizes; (void)n_in; (void)out_size; (void)ws_size;
    const float* X   = (const float*)d_in[0];   // [T, H]
    const float* Wr  = (const float*)d_in[1];   // [H, E]
    const float* Wup = (const float*)d_in[2];   // [E, H, F]
    const float* Wdn = (const float*)d_in[3];   // [E, F, H]
    float* out = (float*)d_out;                 // [T*H] output + [1] aux

    char* p = (char*)d_ws;
    auto take = [&](size_t n) -> char* {
        char* r = p; p += (n + 255) & ~(size_t)255; return r;
    };
    bf16*  Xbf      = (bf16*) take((size_t)T_TOK * H_DIM * 2);
    bf16*  WupT     = (bf16*) take((size_t)NEXP * F_DIM * H_DIM * 2);
    bf16*  WdnT     = (bf16*) take((size_t)NEXP * H_DIM * F_DIM * 2);
    bf16*  Ybf      = (bf16*) take((size_t)MAXSLOTS * H_DIM * 2);
    float* probRow  = (float*)take((size_t)T_TOK * NEXP * 4);
    int*   eidx     = (int*)  take((size_t)NSLOT * 4);
    float* tokw     = (float*)take((size_t)NSLOT * 4);
    int*   tok2slot = (int*)  take((size_t)NSLOT * 4);
    int*   slot2tok = (int*)  take((size_t)MAXSLOTS * 4);
    int*   counts   = (int*)  take(NEXP * 4);
    int*   cursors  = (int*)  take(NEXP * 4);
    int*   tileExpert = (int*)take(MAXTILES * 4);
    int*   tileSlot   = (int*)take(MAXTILES * 4);
    int*   tileValid  = (int*)take(MAXTILES * 4);

    init_kernel<<<(MAXSLOTS + 255) / 256, 256, 0, stream>>>(slot2tok, counts);
    cvtx_kernel<<<T_TOK * H_DIM / 4 / 256, 256, 0, stream>>>(X, Xbf);
    transpose_cvt_kernel<<<dim3(F_DIM / 32, H_DIM / 32, NEXP), dim3(32, 8), 0, stream>>>(
        Wup, WupT, H_DIM, F_DIM);
    transpose_cvt_kernel<<<dim3(H_DIM / 32, F_DIM / 32, NEXP), dim3(32, 8), 0, stream>>>(
        Wdn, WdnT, F_DIM, H_DIM);
    router_kernel<<<T_TOK / 8, 256, 0, stream>>>(X, Wr, probRow, eidx, tokw, counts);
    setup_kernel<<<1, 256, 0, stream>>>(probRow, counts, cursors, tileExpert,
                                        tileSlot, tileValid,
                                        out + (size_t)T_TOK * H_DIM);
    scatter_kernel<<<T_TOK / 256, 256, 0, stream>>>(eidx, cursors, slot2tok, tok2slot);
    moe_gemm_kernel<<<MAXTILES, 256, 0, stream>>>(Xbf, WupT, WdnT, Ybf, slot2tok,
                                                  tileExpert, tileSlot, tileValid);
    combine_kernel<<<T_TOK * H_DIM / 4 / 256, 256, 0, stream>>>(Ybf, tok2slot, tokw, out);
}
